// BahdanauAttention_86741159510307
// MI455X (gfx1250) — matrix operand order repaired and checked by reading, not compiled, not run on hardware
//
#include <hip/hip_runtime.h>
#include <hip/hip_bf16.h>
#include <math.h>

#define B_  16
#define LS_ 1024
#define LH_ 1024
#define D1_ 1024
#define D2_ 1024

typedef __attribute__((ext_vector_type(16))) __bf16 v16bf;
typedef __attribute__((ext_vector_type(8)))  __bf16 v8bf;
typedef __attribute__((ext_vector_type(8)))  float  v8f;
typedef unsigned int u32x4 __attribute__((ext_vector_type(4)));
typedef int          i32x4 __attribute__((ext_vector_type(4)));
typedef int          i32x8 __attribute__((ext_vector_type(8)));

// A-fragment loader: 16-bit A 16x32 lane layout = 8 contiguous + 8 contiguous
// bf16 elements 16 apart (K = kb..kb+7 and kb+16..kb+23). Two 16B loads.
static __device__ __forceinline__ v16bf load_a16(const __bf16* p) {
    v8bf lo = *(const v8bf*)(p);
    v8bf hi = *(const v8bf*)(p + 16);
    return __builtin_shufflevector(lo, hi, 0, 1, 2, 3, 4, 5, 6, 7,
                                           8, 9, 10, 11, 12, 13, 14, 15);
}

// Same A layout but from fp32 source (global or LDS): two float4 pairs + cvt.
static __device__ __forceinline__ v16bf cvt_a16(const float* p) {
    float4 a0 = *(const float4*)(p);
    float4 a1 = *(const float4*)(p + 4);
    float4 b0 = *(const float4*)(p + 16);
    float4 b1 = *(const float4*)(p + 20);
    v16bf r;
    r[0]  = (__bf16)a0.x; r[1]  = (__bf16)a0.y; r[2]  = (__bf16)a0.z; r[3]  = (__bf16)a0.w;
    r[4]  = (__bf16)a1.x; r[5]  = (__bf16)a1.y; r[6]  = (__bf16)a1.z; r[7]  = (__bf16)a1.w;
    r[8]  = (__bf16)b0.x; r[9]  = (__bf16)b0.y; r[10] = (__bf16)b0.z; r[11] = (__bf16)b0.w;
    r[12] = (__bf16)b1.x; r[13] = (__bf16)b1.y; r[14] = (__bf16)b1.z; r[15] = (__bf16)b1.w;
    return r;
}

// B-fragment from 16 contiguous fp32 (K-major source): four float4 + cvt.
static __device__ __forceinline__ v16bf cvt_b16(const float* p) {
    float4 q0 = *(const float4*)(p);
    float4 q1 = *(const float4*)(p + 4);
    float4 q2 = *(const float4*)(p + 8);
    float4 q3 = *(const float4*)(p + 12);
    v16bf r;
    r[0]  = (__bf16)q0.x; r[1]  = (__bf16)q0.y; r[2]  = (__bf16)q0.z; r[3]  = (__bf16)q0.w;
    r[4]  = (__bf16)q1.x; r[5]  = (__bf16)q1.y; r[6]  = (__bf16)q1.z; r[7]  = (__bf16)q1.w;
    r[8]  = (__bf16)q2.x; r[9]  = (__bf16)q2.y; r[10] = (__bf16)q2.z; r[11] = (__bf16)q2.w;
    r[12] = (__bf16)q3.x; r[13] = (__bf16)q3.y; r[14] = (__bf16)q3.z; r[15] = (__bf16)q3.w;
    return r;
}

// ---------------------------------------------------------------------------
// TDM: 2-D tensor load (fp32) global -> LDS. Descriptor per ISA 08 §8.3/8.4:
//   group0: count=1 | lds_addr | global_addr[56:0] | type=2
//   group1: data_size=4B, tensor_dim0/1 = tile_dim0/1, dim0 stride
// Issued by one wave; caller must s_wait_tensorcnt + barrier before use.
// ---------------------------------------------------------------------------
static __device__ __forceinline__ void tdm_load_2d_f32(
    unsigned lds_off, const void* gaddr,
    unsigned tile0, unsigned tile1, unsigned stride0) {
    const unsigned long long ga = (unsigned long long)gaddr;
    u32x4 g0;
    g0[0] = 1u;                                             // count=1 (valid)
    g0[1] = lds_off;                                        // lds_addr
    g0[2] = (unsigned)(ga & 0xFFFFFFFFu);                   // global_addr lo
    g0[3] = (unsigned)((ga >> 32) & 0x01FFFFFFu) | (2u << 30); // ga hi | type=2
    i32x8 g1;
    g1[0] = (int)(2u << 16);                                // data_size=4B
    g1[1] = (int)((tile0 & 0xFFFFu) << 16);                 // tensor_dim0 lo16
    g1[2] = (int)(((tile0 >> 16) & 0xFFFFu) |               // tensor_dim0 hi
                  ((tile1 & 0xFFFFu) << 16));               // tensor_dim1 lo
    g1[3] = (int)(((tile1 >> 16) & 0xFFFFu) |               // tensor_dim1 hi
                  ((tile0 & 0xFFFFu) << 16));               // tile_dim0
    g1[4] = (int)(tile1 & 0xFFFFu);                         // tile_dim1, tile_dim2=0
    g1[5] = (int)stride0;                                   // dim0 stride lo32
    g1[6] = 0;                                              // stride0 hi | stride1 lo
    g1[7] = 0;
    i32x4 z4 = {0, 0, 0, 0};
#if __clang_major__ >= 23
    i32x8 z8 = {0, 0, 0, 0, 0, 0, 0, 0};
    __builtin_amdgcn_tensor_load_to_lds(g0, g1, z4, z4, z8, 0);
#else
    __builtin_amdgcn_tensor_load_to_lds(g0, g1, z4, z4, 0);
#endif
}

// ---------------------------------------------------------------------------
// Kernel 1: H (fp32) -> Hb (bf16, row-major [b][t][e]) and
//                       HbT (bf16, transposed  [b][e][t])
// 64x64 tile through padded LDS so both writes are coalesced.
// ---------------------------------------------------------------------------
__global__ __launch_bounds__(256)
void h_convert_kernel(const float* __restrict__ H, __bf16* __restrict__ Hb,
                      __bf16* __restrict__ HbT) {
    __shared__ float tile[64][65];
    const int b  = blockIdx.z;
    const int tB = blockIdx.y * 64;      // t tile base
    const int eB = blockIdx.x * 64;      // e tile base
    const int tx = threadIdx.x & 63;
    const int ty = threadIdx.x >> 6;     // 0..3
    #pragma unroll 4
    for (int r = 0; r < 16; ++r) {
        const int t = tB + ty + 4 * r;
        const float v = H[((size_t)b * LH_ + t) * D2_ + eB + tx];
        tile[ty + 4 * r][tx] = v;
        Hb[((size_t)b * LH_ + t) * D2_ + eB + tx] = (__bf16)v;
    }
    __syncthreads();
    #pragma unroll 4
    for (int r = 0; r < 16; ++r) {
        const int e = eB + ty + 4 * r;
        HbT[((size_t)b * D2_ + e) * LH_ + tB + tx] = (__bf16)tile[tx][ty + 4 * r];
    }
}

// ---------------------------------------------------------------------------
// Kernel 2: projection GEMM  S_[m,e] = sum_d S[m,d]*W[e,d] + bias[e]
//   M = B*LS = 16384, N = D2 = 1024, K = D1 = 1024.
//   The 16-row A block (64 KB fp32) is shared by all 8 waves -> staged into
//   LDS once per workgroup via the Tensor Data Mover, then each wave reads
//   its A-fragments from LDS while streaming B (= rows of W) from global.
// ---------------------------------------------------------------------------
__global__ __launch_bounds__(256)
void proj_gemm_kernel(const float* __restrict__ S, const float* __restrict__ W,
                      const float* __restrict__ bias, __bf16* __restrict__ Sb) {
    __shared__ float As[16 * D1_];       // 64 KB staged A rows (TDM dest)
    const int lane  = threadIdx.x & 31;
    const int wave  = threadIdx.x >> 5;
    const int half  = lane >> 4;
    const int l16   = lane & 15;
    const int mBase = blockIdx.y << 4;
    const int nBase = (blockIdx.x << 7) + (wave << 4);

    if (wave == 0) {                     // one wave drives the TDM
        tdm_load_2d_f32((unsigned)(uintptr_t)As,
                        S + (size_t)mBase * D1_,
                        /*tile0=*/D1_, /*tile1=*/16, /*stride0=*/D1_);
        __builtin_amdgcn_s_wait_tensorcnt(0);
    }
    __syncthreads();                     // LDS block visible to all waves

    const float* Ap = As + (size_t)l16 * D1_ + half * 8;          // from LDS
    const float* Bp = W + (size_t)(nBase + l16) * D1_ + half * 8;  // from HBM

    v8f c = {};
    for (int kt = 0; kt < D1_ / 32; ++kt) {
        const int d0 = kt * 32;
        __builtin_prefetch(Bp + d0 + 64, 0, 0);   // next-next K tile -> L2
        v16bf a = cvt_a16(Ap + d0);
        v16bf b = cvt_a16(Bp + d0);
        c = __builtin_amdgcn_wmma_f32_16x16x32_bf16(
                false, a, false, b, (short)0, c, false, false);
    }
    const float bv = bias[nBase + l16];
    #pragma unroll
    for (int i = 0; i < 8; ++i) {        // C: VGPR i -> M = i + 8*half
        const int r = mBase + half * 8 + i;
        Sb[(size_t)r * D2_ + nBase + l16] = (__bf16)(c[i] + bv);
    }
}

// ---------------------------------------------------------------------------
// Kernel 3: fused attention per (batch, 16-row s-block).
//   Phase 1: score[16][1024] via WMMA into LDS (fp32), B from Hb (contig)
//   Phase 2: pad-mask add + softmax (wave32 shuffle reductions)
//   Phase 3: out = P @ H via WMMA, A from LDS probs, B from HbT (contig)
// ---------------------------------------------------------------------------
__global__ __launch_bounds__(256)
void attn_kernel(const __bf16* __restrict__ Sb, const __bf16* __restrict__ Hb,
                 const __bf16* __restrict__ HbT, const float* __restrict__ pad,
                 float* __restrict__ out) {
    __shared__ float sc[16 * LH_];       // 64 KB score/prob block
    const int lane  = threadIdx.x & 31;
    const int wave  = threadIdx.x >> 5;
    const int half  = lane >> 4;
    const int l16   = lane & 15;
    const int b     = blockIdx.x >> 6;
    const int sBase = (blockIdx.x & 63) << 4;

    // ----- Phase 1: score[s,t] = sum_e S_[s,e] * H[t,e] -----
    const __bf16* Ap = Sb + (size_t)(b * LS_ + sBase + l16) * D2_ + half * 8;
    for (int tt = wave; tt < LH_ / 16; tt += 8) {
        v8f c = {};
        const int tBase = tt << 4;
        const __bf16* Bp = Hb + (size_t)(b * LH_ + tBase + l16) * D2_ + half * 8;
        for (int kt = 0; kt < D2_ / 32; ++kt) {
            const int e0 = kt * 32;
            __builtin_prefetch(Bp + e0 + 64, 0, 0);
            v16bf a  = load_a16(Ap + e0);
            v16bf bb = load_a16(Bp + e0);
            c = __builtin_amdgcn_wmma_f32_16x16x32_bf16(
                    false, a, false, bb, (short)0, c, false, false);
        }
        #pragma unroll
        for (int i = 0; i < 8; ++i)
            sc[(half * 8 + i) * LH_ + tBase + l16] = c[i];
    }
    __syncthreads();

    // ----- Phase 2: pad mask + softmax over t (one row per half-wave) -----
    {
        const int row = wave * 2 + half;                // 0..15
        float* rp = sc + row * LH_;
        const float* pm = pad + ((size_t)b * LS_ + sBase + row) * LH_;
        float mx = -3.402823466e38f;
        for (int j = 0; j < LH_ / 16; ++j) {
            const int col = l16 + j * 16;
            const float v = rp[col] + pm[col];
            rp[col] = v;
            mx = fmaxf(mx, v);
        }
        mx = fmaxf(mx, __shfl_xor(mx, 1));
        mx = fmaxf(mx, __shfl_xor(mx, 2));
        mx = fmaxf(mx, __shfl_xor(mx, 4));
        mx = fmaxf(mx, __shfl_xor(mx, 8));
        float sum = 0.f;
        for (int j = 0; j < LH_ / 16; ++j) {
            const int col = l16 + j * 16;
            const float v = __expf(rp[col] - mx);
            rp[col] = v;
            sum += v;
        }
        sum += __shfl_xor(sum, 1);
        sum += __shfl_xor(sum, 2);
        sum += __shfl_xor(sum, 4);
        sum += __shfl_xor(sum, 8);
        const float inv = 1.f / sum;
        for (int j = 0; j < LH_ / 16; ++j)
            rp[l16 + j * 16] *= inv;
    }
    __syncthreads();

    // ----- Phase 3: out[s,e] = sum_t P[s,t] * H[t,e] -----
    const float* Pp = sc + l16 * LH_ + half * 8;        // A = prob tile rows
    for (int et = wave; et < D2_ / 16; et += 8) {
        v8f c = {};
        const int ecol = (et << 4) + l16;
        const __bf16* Bp = HbT + ((size_t)b * D2_ + ecol) * LH_ + half * 8;
        for (int kt = 0; kt < LH_ / 32; ++kt) {
            const int t0 = kt * 32;
            __builtin_prefetch(Bp + t0 + 64, 0, 0);
            v16bf a  = cvt_a16(Pp + t0);                // f32 LDS -> bf16
            v16bf bb = load_a16(Bp + t0);               // contiguous via HbT
            c = __builtin_amdgcn_wmma_f32_16x16x32_bf16(
                    false, a, false, bb, (short)0, c, false, false);
        }
        #pragma unroll
        for (int i = 0; i < 8; ++i)
            out[((size_t)b * LS_ + sBase + half * 8 + i) * (size_t)D2_ + ecol]
                = c[i];
    }
}

// ---------------------------------------------------------------------------
extern "C" void kernel_launch(void* const* d_in, const int* in_sizes, int n_in,
                              void* d_out, int out_size, void* d_ws, size_t ws_size,
                              hipStream_t stream) {
    const float* S   = (const float*)d_in[0];   // [B, LS, D1]
    const float* H   = (const float*)d_in[1];   // [B, LH, D2]
    const float* pad = (const float*)d_in[2];   // [B, LS, LH]
    const float* Ww  = (const float*)d_in[3];   // [D2, D1]
    const float* Wb  = (const float*)d_in[4];   // [D2]
    float* out = (float*)d_out;                 // [B, LS, D2]

    // workspace: S_ bf16 (32MB) | H bf16 (32MB) | H^T bf16 (32MB)
    const size_t seg = (size_t)B_ * LH_ * D2_ * sizeof(__bf16);
    __bf16* Sbw = (__bf16*)d_ws;
    __bf16* Hb  = (__bf16*)((char*)d_ws + seg);
    __bf16* HbT = (__bf16*)((char*)d_ws + 2 * seg);

    h_convert_kernel<<<dim3(D2_ / 64, LH_ / 64, B_), 256, 0, stream>>>(H, Hb, HbT);
    proj_gemm_kernel<<<dim3(D2_ / 128, (B_ * LS_) / 16), 256, 0, stream>>>(S, Ww, Wb, Sbw);
    attn_kernel<<<B_ * (LS_ / 16), 256, 0, stream>>>(Sbw, Hb, HbT, pad, out);
}
